// GaussianRasterizer_61091614818789
// MI455X (gfx1250) — compile-verified, matrix operand order checked by reading, not run on hardware
//
#include <hip/hip_runtime.h>
#include <hip/hip_bf16.h>

typedef __attribute__((ext_vector_type(16))) _Float16 v16h;
typedef __attribute__((ext_vector_type(8)))  _Float16 v8h;
typedef __attribute__((ext_vector_type(8)))  float    v8f;

#define NG   1024          // gaussians
#define FD   512           // feature dim
#define OD   512           // output channels
#define IMW  128
#define IMH  128
#define NPIX (IMW*IMH)     // 16384

// ---------------------------------------------------------------------------
// 1) Per-gaussian preprocessing: projection, conic, radii, depth
// ---------------------------------------------------------------------------
__global__ void gs_preprocess(const float* __restrict__ means3D,
                              const float* __restrict__ opac,
                              const float* __restrict__ scales,
                              const float* __restrict__ rots,
                              const float* __restrict__ vm,
                              const float* __restrict__ pm,
                              float4* __restrict__ preA,
                              float4* __restrict__ preB,
                              float*  __restrict__ depthArr,
                              float*  __restrict__ radii_out) {
  int n = blockIdx.x * blockDim.x + threadIdx.x;
  if (n >= NG) return;
  float ph[4] = { means3D[n*3], means3D[n*3+1], means3D[n*3+2], 1.f };
  float pv[4], phom[4];
  #pragma unroll
  for (int j = 0; j < 4; j++) {
    float a = 0.f, b = 0.f;
    #pragma unroll
    for (int k = 0; k < 4; k++) { a += ph[k]*vm[k*4+j]; b += ph[k]*pm[k*4+j]; }
    pv[j] = a; phom[j] = b;
  }
  float inw = 1.f / (phom[3] + 1e-7f);
  float px = ((phom[0]*inw + 1.f) * (float)IMW - 1.f) * 0.5f;
  float py = ((phom[1]*inw + 1.f) * (float)IMH - 1.f) * 0.5f;
  float depth = pv[2];

  // quaternion -> rotation
  float q0 = rots[n*4], q1 = rots[n*4+1], q2 = rots[n*4+2], q3 = rots[n*4+3];
  float qn = rsqrtf(q0*q0 + q1*q1 + q2*q2 + q3*q3);
  float r = q0*qn, x = q1*qn, y = q2*qn, z = q3*qn;
  float R[3][3] = {
    {1.f-2.f*(y*y+z*z), 2.f*(x*y-r*z),     2.f*(x*z+r*y)},
    {2.f*(x*y+r*z),     1.f-2.f*(x*x+z*z), 2.f*(y*z-r*x)},
    {2.f*(x*z-r*y),     2.f*(y*z+r*x),     1.f-2.f*(x*x+y*y)}};
  float s[3] = { scales[n*3], scales[n*3+1], scales[n*3+2] };
  float C3[3][3];
  #pragma unroll
  for (int i = 0; i < 3; i++)
    #pragma unroll
    for (int j = 0; j < 3; j++) {
      float a = 0.f;
      #pragma unroll
      for (int k = 0; k < 3; k++) a += R[i][k]*R[j][k]*s[k]*s[k];
      C3[i][j] = a;
    }
  const float fx = 128.f, fy = 128.f, lim = 1.3f*0.5f;
  float tz = pv[2], itz = 1.f/tz;
  float txz = fminf(fmaxf(pv[0]*itz, -lim), lim) * tz;
  float tyz = fminf(fmaxf(pv[1]*itz, -lim), lim) * tz;
  float J[2][3] = { {fx*itz, 0.f, -fx*txz*itz*itz},
                    {0.f, fy*itz, -fy*tyz*itz*itz} };
  float Tm[2][3];
  #pragma unroll
  for (int a = 0; a < 2; a++)
    #pragma unroll
    for (int b = 0; b < 3; b++) {
      float acc = 0.f;
      #pragma unroll
      for (int c = 0; c < 3; c++) acc += J[a][c]*vm[b*4+c];
      Tm[a][b] = acc;
    }
  float c2[2][2];
  #pragma unroll
  for (int a = 0; a < 2; a++)
    #pragma unroll
    for (int b = 0; b < 2; b++) {
      float acc = 0.f;
      #pragma unroll
      for (int i = 0; i < 3; i++)
        #pragma unroll
        for (int j = 0; j < 3; j++) acc += Tm[a][i]*C3[i][j]*Tm[b][j];
      c2[a][b] = acc;
    }
  c2[0][0] += 0.3f; c2[1][1] += 0.3f;
  float det  = c2[0][0]*c2[1][1] - c2[0][1]*c2[0][1];
  float dets = (det == 0.f) ? 1.f : det;
  float id   = 1.f / dets;
  float cA = c2[1][1]*id, cB = -c2[0][1]*id, cC = c2[0][0]*id;
  float mid = 0.5f*(c2[0][0] + c2[1][1]);
  float lam1 = mid + sqrtf(fmaxf(0.1f, mid*mid - det));
  int rad = (int)ceilf(3.f*sqrtf(lam1));
  bool valid = (depth > 0.2f) && (det != 0.f);
  if (!valid) rad = 0;

  preA[n] = make_float4(px, py, cA, cB);
  preB[n] = make_float4(cC, opac[n], valid ? 1.f : 0.f, __int_as_float(n));
  depthArr[n] = depth;
  radii_out[n] = (float)rad;
}

// ---------------------------------------------------------------------------
// 2) Stable rank sort by depth (N=1024, one workgroup)
// ---------------------------------------------------------------------------
__global__ void gs_sort(const float* __restrict__ depth, int* __restrict__ order) {
  __shared__ float sd[NG];
  int n = threadIdx.x;
  float d = depth[n];
  sd[n] = d;
  __syncthreads();
  int rank = 0;
  for (int m = 0; m < NG; m++) {
    float dm = sd[m];
    rank += (dm < d) || (dm == d && m < n);
  }
  order[rank] = n;
}

// ---------------------------------------------------------------------------
// 3) Gather packed per-gaussian params in depth order (3 float4 / gaussian)
// ---------------------------------------------------------------------------
__global__ void gs_gather(const int* __restrict__ order,
                          const float4* __restrict__ preA,
                          const float4* __restrict__ preB,
                          const float*  __restrict__ colors,
                          float4* __restrict__ sp) {
  int i = blockIdx.x * blockDim.x + threadIdx.x;
  if (i >= NG) return;
  int n = order[i];
  sp[i*3+0] = preA[n];
  sp[i*3+1] = preB[n];
  sp[i*3+2] = make_float4(colors[n*3], colors[n*3+1], colors[n*3+2], 0.f);
}

// ---------------------------------------------------------------------------
// 4) Per-pixel compositing: weights W^T[pixel][n] (f16), color, transmittance
// ---------------------------------------------------------------------------
__global__ void gs_composite(const float4* __restrict__ sp,
                             const float*  __restrict__ bg,
                             _Float16* __restrict__ WT,
                             float* __restrict__ outColor) {
  int p = blockIdx.x * blockDim.x + threadIdx.x;   // 16384 pixels
  float gx = (float)(p & (IMW-1));
  float gy = (float)(p >> 7);
  float T = 1.f, cr = 0.f, cg = 0.f, cb = 0.f;
  _Float16* wrow = WT + (size_t)p * NG;
  for (int i = 0; i < NG; i++) {
    float4 a = sp[i*3+0];
    float4 b = sp[i*3+1];
    float4 c = sp[i*3+2];
    float dx = a.x - gx, dy = a.y - gy;
    float power = -0.5f*(a.z*dx*dx + b.x*dy*dy) - a.w*dx*dy;
    float alpha = fminf(0.99f, b.y * __expf(power));
    bool keep = (power <= 0.f) && (alpha >= (1.f/255.f)) && (b.z > 0.5f);
    float w = keep ? T*alpha : 0.f;
    if (keep) T *= (1.f - alpha);
    wrow[__float_as_int(b.w)] = (_Float16)w;       // scatter by original index
    cr += w*c.x; cg += w*c.y; cb += w*c.z;
  }
  outColor[p]          = cr + T*bg[0];
  outColor[NPIX + p]   = cg + T*bg[1];
  outColor[2*NPIX + p] = cb + T*bg[2];
}

// ---------------------------------------------------------------------------
// 5) f32 -> f16 conversion
// ---------------------------------------------------------------------------
__global__ void f32_to_f16(const float* __restrict__ src, _Float16* __restrict__ dst, int n) {
  int i = blockIdx.x * blockDim.x + threadIdx.x;
  if (i < n) dst[i] = (_Float16)src[i];
}

// ---------------------------------------------------------------------------
// 6) WMMA GEMM:  D[M,N] = A[M,K] * Bt[N,K]^T  (+ bias), f16 in / f32 acc.
//    One wave per 16x64 strip: A fragment reused across 4 N-tiles
//    (4 WMMAs per A-fragment load -> 2.5 vmem per WMMA instead of 4).
// ---------------------------------------------------------------------------
__device__ __forceinline__ v16h ld_frag(const _Float16* p, int K) {
  // halves [0..7]  = K+0..7 of this half-wave's chunk
  // halves [8..15] = K+16..23 of this half-wave's chunk (per ISA A/B layout)
  v8h c0 = *(const v8h*)(p);
  v8h c1 = *(const v8h*)(p + 16);
  v16h v;
  #pragma unroll
  for (int i = 0; i < 8; i++) { v[i] = c0[i]; v[i+8] = c1[i]; }
  (void)K;
  return v;
}

__global__ void gemm_wmma_f16(const _Float16* __restrict__ A,
                              const _Float16* __restrict__ Bt,
                              _Float16* __restrict__ D16,   // f16 out (or null)
                              float*    __restrict__ D32,   // f32 out (or null)
                              const float* __restrict__ bias,
                              int M, int Nn, int K) {
  int wid  = (blockIdx.x * blockDim.x + threadIdx.x) >> 5;
  int lane = threadIdx.x & 31;
  int groupsN = Nn >> 6;                    // groups of 4 N-tiles (64 cols)
  int tm = wid / groupsN;
  int tg = wid - tm * groupsN;
  if (tm * 16 >= M) return;                 // uniform per wave

  int mr    = lane & 15;                    // row / col within tile
  int khalf = (lane >> 4) * 8;              // K sub-chunk per half-wave
  const _Float16* ap = A  + (size_t)(tm*16 + mr) * K + khalf;
  const _Float16* bp = Bt + (size_t)(tg*64 + mr) * K + khalf;
  size_t bstep = (size_t)16 * K;            // 16 B-rows (one N-tile) apart

  v8f acc0 = {}, acc1 = {}, acc2 = {}, acc3 = {};
  for (int k = 0; k < K; k += 32) {
    v16h av = ld_frag(ap + k, K);
    v16h b0 = ld_frag(bp + k, K);
    v16h b1 = ld_frag(bp + bstep   + k, K);
    v16h b2 = ld_frag(bp + 2*bstep + k, K);
    v16h b3 = ld_frag(bp + 3*bstep + k, K);
    acc0 = __builtin_amdgcn_wmma_f32_16x16x32_f16(false, av, false, b0, (short)0, acc0, false, false);
    acc1 = __builtin_amdgcn_wmma_f32_16x16x32_f16(false, av, false, b1, (short)0, acc1, false, false);
    acc2 = __builtin_amdgcn_wmma_f32_16x16x32_f16(false, av, false, b2, (short)0, acc2, false, false);
    acc3 = __builtin_amdgcn_wmma_f32_16x16x32_f16(false, av, false, b3, (short)0, acc3, false, false);
  }

  int rbase = tm*16 + ((lane >> 4) << 3);   // acc element r -> row rbase+r
  v8f accs[4] = { acc0, acc1, acc2, acc3 };
  #pragma unroll
  for (int t = 0; t < 4; t++) {
    int col = tg*64 + t*16 + mr;
    if (D16) {
      #pragma unroll
      for (int r = 0; r < 8; r++)
        D16[(size_t)(rbase + r) * Nn + col] = (_Float16)accs[t][r];
    } else {
      #pragma unroll
      for (int r = 0; r < 8; r++)
        D32[(size_t)(rbase + r) * Nn + col] = accs[t][r] + bias[rbase + r];
    }
  }
}

// ---------------------------------------------------------------------------
// Launch
// ---------------------------------------------------------------------------
extern "C" void kernel_launch(void* const* d_in, const int* in_sizes, int n_in,
                              void* d_out, int out_size, void* d_ws, size_t ws_size,
                              hipStream_t stream) {
  const float* means3D = (const float*)d_in[0];
  const float* opac    = (const float*)d_in[2];
  const float* colors  = (const float*)d_in[3];
  const float* feat    = (const float*)d_in[4];
  const float* scales  = (const float*)d_in[5];
  const float* rots    = (const float*)d_in[6];
  const float* vm      = (const float*)d_in[7];
  const float* pm      = (const float*)d_in[8];
  const float* bg      = (const float*)d_in[9];
  const float* convw   = (const float*)d_in[11];
  const float* convb   = (const float*)d_in[12];
  float* out = (float*)d_out;
  float* out_feat  = out + 3*NPIX;               // 512x128x128
  float* out_radii = out + 3*NPIX + (size_t)OD*NPIX;

  // workspace layout
  char* ws = (char*)d_ws;
  float4*   preA    = (float4*)(ws);                      //  16 KB
  float4*   preB    = (float4*)(ws + 16384);              //  16 KB
  float*    depth   = (float*) (ws + 32768);              //   4 KB
  int*      order   = (int*)   (ws + 36864);              //   4 KB
  float4*   sp      = (float4*)(ws + 40960);              //  48 KB
  _Float16* convw16 = (_Float16*)(ws + 90112);            // 512 KB
  _Float16* feat16  = (_Float16*)(ws + 614400);           //   1 MB
  _Float16* G16     = (_Float16*)(ws + 1662976);          //   1 MB  (OD x NG)
  _Float16* WT      = (_Float16*)(ws + 2711552);          //  32 MB  (NPIX x NG)

  // 1) preprocess
  gs_preprocess<<<(NG+255)/256, 256, 0, stream>>>(means3D, opac, scales, rots,
                                                  vm, pm, preA, preB, depth, out_radii);
  // 2) depth sort
  gs_sort<<<1, NG, 0, stream>>>(depth, order);
  // 3) gather sorted packed params
  gs_gather<<<(NG+255)/256, 256, 0, stream>>>(order, preA, preB, colors, sp);
  // 4) per-pixel composite -> WT (f16), color
  gs_composite<<<NPIX/256, 256, 0, stream>>>(sp, bg, WT, out);
  // 5) converts
  f32_to_f16<<<(OD*FD+255)/256, 256, 0, stream>>>(convw, convw16, OD*FD);
  f32_to_f16<<<(NG*FD+255)/256, 256, 0, stream>>>(feat,  feat16,  NG*FD);
  // 6) G = conv_w(512x512) * feat^T(512x1024)  -> f16 [OD][NG]
  {
    int waves = (OD/16) * (NG/64);               // 32 * 16 = 512
    gemm_wmma_f16<<<waves/8, 256, 0, stream>>>(convw16, feat16, G16, nullptr,
                                               nullptr, OD, NG, FD);
  }
  // 7) feature_out = G(512x1024) * W(1024x16384) + b  -> f32 [OD][NPIX]
  {
    int waves = (OD/16) * (NPIX/64);             // 32 * 256 = 8192
    gemm_wmma_f16<<<waves/8, 256, 0, stream>>>(G16, WT, nullptr, out_feat,
                                               convb, OD, NPIX, NG);
  }
}
